// LSTM_10522669875315
// MI455X (gfx1250) — matrix operand order repaired and checked by reading, not compiled, not run on hardware
//
#include <hip/hip_runtime.h>

#define BATCH 256
#define TLEN  256
#define HID   1024
#define EMB   10
#define NVOC  10
#define NCLS  10

typedef __attribute__((ext_vector_type(16))) __bf16 v16bf;
typedef __attribute__((ext_vector_type(8)))  float  v8f;

__device__ __forceinline__ unsigned short f2bf(float f) {
    unsigned u = __float_as_uint(f);
    u += 0x7FFFu + ((u >> 16) & 1u);      // round-to-nearest-even
    return (unsigned short)(u >> 16);
}
__device__ __forceinline__ float sigm(float x) { return 1.0f / (1.0f + __expf(-x)); }
// tanh(x) = 2*sigmoid(2x) - 1 : ~5 VALU ops instead of branchy libm tanhf
__device__ __forceinline__ float tanh_fast(float x) {
    return __builtin_fmaf(2.0f, sigm(2.0f * x), -1.0f);
}

// ---- Convert the 4 recurrent weight matrices to bf16, stacked [gate][j][k] ----
__global__ void conv_wh(const float* __restrict__ Wg, const float* __restrict__ Wi,
                        const float* __restrict__ Wf, const float* __restrict__ Wo,
                        unsigned short* __restrict__ Whb) {
    int idx = blockIdx.x * blockDim.x + threadIdx.x;           // 4*H*H threads
    int g   = idx >> 20;
    int off = idx & ((1 << 20) - 1);
    const float* src = (g == 0) ? Wg : (g == 1) ? Wi : (g == 2) ? Wf : Wo;
    Whb[idx] = f2bf(src[off]);
}

// ---- proj[v][q*H+j] = b_q[j] + sum_e embed[v][e] * W_qx[j][e]  (token lookup table) ----
__global__ void build_proj(const float* __restrict__ embed,
                           const float* __restrict__ Wgx, const float* __restrict__ Wix,
                           const float* __restrict__ Wfx, const float* __restrict__ Wox,
                           const float* __restrict__ bg,  const float* __restrict__ bi,
                           const float* __restrict__ bf_, const float* __restrict__ bo,
                           float* __restrict__ proj) {
    int idx = blockIdx.x * blockDim.x + threadIdx.x;           // NVOC*4*H
    if (idx >= NVOC * 4 * HID) return;
    int v   = idx / (4 * HID);
    int rem = idx % (4 * HID);
    int q   = rem / HID;
    int j   = rem % HID;
    const float* W  = (q == 0) ? Wgx : (q == 1) ? Wix : (q == 2) ? Wfx : Wox;
    const float* bb = (q == 0) ? bg  : (q == 1) ? bi  : (q == 2) ? bf_ : bo;
    float acc = bb[j];
    #pragma unroll
    for (int e = 0; e < EMB; e++) acc += embed[v * EMB + e] * W[j * EMB + e];
    proj[v * 4 * HID + q * HID + j] = acc;
}

__global__ void init_state(float* __restrict__ c, unsigned short* __restrict__ h0,
                           float* __restrict__ hf) {
    int idx = blockIdx.x * blockDim.x + threadIdx.x;           // BATCH*HID
    c[idx]  = 0.0f;
    h0[idx] = 0;
    hf[idx] = 0.0f;
}

// ---- One LSTM timestep: z = proj[tok] + h @ Wh^T (bf16 WMMA), fused gate math ----
// Double-buffered pipeline using gfx1250 async global->LDS loads (ASYNCcnt).
__global__ __launch_bounds__(128)
void lstm_step(const unsigned short* __restrict__ Whb,   // [4][H][H] bf16
               const float*          __restrict__ proj,  // [NVOC][4H]
               const int*            __restrict__ x,     // [BATCH][TLEN]
               int t,
               const unsigned short* __restrict__ hprev, // [BATCH][H] bf16
               unsigned short*       __restrict__ hnext, // [BATCH][H] bf16
               float*                __restrict__ cbuf,  // [BATCH][H]
               float*                __restrict__ hf)    // [BATCH][H] fp32
{
    // K-chunk = 64. Row stride 72 halfs (144 B) keeps every b128 access 16B aligned.
    __shared__ alignas(16) unsigned short Ab[2][32][72];       // h tile: [m][k]
    __shared__ alignas(16) unsigned short Bb[2][4][32][72];    // per-gate Wh tile: [n][k]
    __shared__ int toks[32];

    const int tid  = threadIdx.x;
    const int lane = tid & 31;
    const int wave = tid >> 5;
    const int hi   = lane >> 4;       // lane half (K split per ISA layout)
    const int l15  = lane & 15;
    const int mw   = wave >> 1;       // wave's 16x16 subtile row
    const int nw   = wave & 1;        // wave's 16x16 subtile col
    const int m0   = blockIdx.y * 32; // batch tile base
    const int n0   = blockIdx.x * 32; // hidden tile base

    if (tid < 32) toks[tid] = x[(m0 + tid) * TLEN + t];

    v8f acc[4];
    #pragma unroll
    for (int g = 0; g < 4; g++)
        #pragma unroll
        for (int v = 0; v < 8; v++) acc[g][v] = 0.0f;

    // Issue the 10 async b128 loads (per thread) that fill one K-chunk of A+B.
    auto issue_chunk = [&](int kc, int buf) {
        #pragma unroll
        for (int p = 0; p < 2; p++) {                 // A: 256 x 16B transactions
            int idx = tid + 128 * p;
            int row = idx >> 3;
            int ck  = idx & 7;
            unsigned lds = (unsigned)(uintptr_t)&Ab[buf][row][ck * 8];
            unsigned long long ga = (unsigned long long)(uintptr_t)
                (hprev + (size_t)(m0 + row) * HID + kc + ck * 8);
            asm volatile("global_load_async_to_lds_b128 %0, %1, off"
                         :: "v"(lds), "v"(ga) : "memory");
        }
        #pragma unroll
        for (int p = 0; p < 8; p++) {                 // B: 1024 x 16B transactions
            int idx = tid + 128 * p;
            int g   = idx >> 8;
            int rem = idx & 255;
            int row = rem >> 3;
            int ck  = rem & 7;
            unsigned lds = (unsigned)(uintptr_t)&Bb[buf][g][row][ck * 8];
            unsigned long long ga = (unsigned long long)(uintptr_t)
                (Whb + (size_t)g * HID * HID + (size_t)(n0 + row) * HID + kc + ck * 8);
            asm volatile("global_load_async_to_lds_b128 %0, %1, off"
                         :: "v"(lds), "v"(ga) : "memory");
        }
    };

    issue_chunk(0, 0);

    int buf = 0;
    for (int kc = 0; kc < HID; kc += 64, buf ^= 1) {
        if (kc + 64 < HID) {
            issue_chunk(kc + 64, buf ^ 1);            // overlap next chunk's loads
            asm volatile("s_wait_asynccnt 10" ::: "memory");  // chunk kc complete
        } else {
            asm volatile("s_wait_asynccnt 0" ::: "memory");
        }
        __syncthreads();                              // all waves' LDS data visible

        #pragma unroll
        for (int ks = 0; ks < 2; ks++) {              // two K=32 slices
            // A fragment 16x32: lanes 0-15 -> K {0..7,16..23}; lanes 16-31 -> K {8..15,24..31}
            union { v16bf v; uint4 q[2]; } afr;
            afr.q[0] = *(const uint4*)&Ab[buf][16 * mw + l15][ks * 32 + hi * 8];
            afr.q[1] = *(const uint4*)&Ab[buf][16 * mw + l15][ks * 32 + 16 + hi * 8];
            #pragma unroll
            for (int g = 0; g < 4; g++) {
                // B fragment 32x16: lanes 0-15 -> K {0..7,16..23}; lanes 16-31 -> K {8..15,24..31}
                union { v16bf v; uint4 q[2]; } bfr;
                bfr.q[0] = *(const uint4*)&Bb[buf][g][16 * nw + l15][ks * 32 + hi * 8];
                bfr.q[1] = *(const uint4*)&Bb[buf][g][16 * nw + l15][ks * 32 + 16 + hi * 8];
                acc[g] = __builtin_amdgcn_wmma_f32_16x16x32_bf16(
                             false, afr.v, false, bfr.v, (short)0, acc[g], false, false);
            }
        }
        __syncthreads();                              // reads done before buf is re-filled
    }

    // Fused gate math + state update. C/D layout: element v of lane -> row v + 8*hi, col l15.
    const int ng = n0 + 16 * nw + l15;                // global hidden column
    #pragma unroll
    for (int v = 0; v < 8; v++) {
        int mloc = 16 * mw + 8 * hi + v;              // local batch row in tile
        int m    = m0 + mloc;
        const float* pr = proj + (size_t)toks[mloc] * (4 * HID);
        float zg = acc[0][v] + pr[ng];
        float zi = acc[1][v] + pr[HID + ng];
        float zf = acc[2][v] + pr[2 * HID + ng];
        float zo = acc[3][v] + pr[3 * HID + ng];
        float gg = tanh_fast(zg);
        float ii = sigm(zi);
        float ff = sigm(zf);
        float oo = sigm(zo);
        size_t idx = (size_t)m * HID + ng;
        float cn = gg * ii + cbuf[idx] * ff;
        cbuf[idx] = cn;
        float hn = tanh_fast(cn) * oo;
        hf[idx]    = hn;
        hnext[idx] = f2bf(hn);
    }
}

// ---- y = h @ W_ph^T + b_p  (tiny, fp32) ----
__global__ void final_proj(const float* __restrict__ hf, const float* __restrict__ Wph,
                           const float* __restrict__ bp, float* __restrict__ out) {
    int idx = blockIdx.x * blockDim.x + threadIdx.x;
    if (idx >= BATCH * NCLS) return;
    int b = idx / NCLS, cl = idx % NCLS;
    float acc = bp[cl];
    const float* hp = hf + (size_t)b * HID;
    const float* wp = Wph + (size_t)cl * HID;
    for (int k = 0; k < HID; k++) acc += hp[k] * wp[k];
    out[idx] = acc;
}

extern "C" void kernel_launch(void* const* d_in, const int* in_sizes, int n_in,
                              void* d_out, int out_size, void* d_ws, size_t ws_size,
                              hipStream_t stream) {
    const int*   x     = (const int*)d_in[0];
    const float* embed = (const float*)d_in[1];
    const float* Wgx = (const float*)d_in[2];
    const float* Wix = (const float*)d_in[3];
    const float* Wfx = (const float*)d_in[4];
    const float* Wox = (const float*)d_in[5];
    const float* Wgh = (const float*)d_in[6];
    const float* Wih = (const float*)d_in[7];
    const float* Wfh = (const float*)d_in[8];
    const float* Woh = (const float*)d_in[9];
    const float* Wph = (const float*)d_in[10];
    const float* bg  = (const float*)d_in[11];
    const float* bi  = (const float*)d_in[12];
    const float* bf  = (const float*)d_in[13];
    const float* bo  = (const float*)d_in[14];
    const float* bp  = (const float*)d_in[15];

    char* ws = (char*)d_ws;
    size_t off = 0;
    unsigned short* Whb  = (unsigned short*)(ws + off); off += (size_t)4 * HID * HID * 2; // 8 MB
    float*          proj = (float*)(ws + off);          off += (size_t)NVOC * 4 * HID * 4;
    unsigned short* h0   = (unsigned short*)(ws + off); off += (size_t)BATCH * HID * 2;
    unsigned short* h1   = (unsigned short*)(ws + off); off += (size_t)BATCH * HID * 2;
    float*          cbuf = (float*)(ws + off);          off += (size_t)BATCH * HID * 4;
    float*          hfb  = (float*)(ws + off);          off += (size_t)BATCH * HID * 4;

    conv_wh<<<(4 * HID * HID) / 256, 256, 0, stream>>>(Wgh, Wih, Wfh, Woh, Whb);
    build_proj<<<(NVOC * 4 * HID + 255) / 256, 256, 0, stream>>>(
        embed, Wgx, Wix, Wfx, Wox, bg, bi, bf, bo, proj);
    init_state<<<(BATCH * HID) / 256, 256, 0, stream>>>(cbuf, h0, hfb);

    for (int t = 0; t < TLEN; t++) {
        const unsigned short* hp = (t & 1) ? h1 : h0;
        unsigned short*       hn = (t & 1) ? h0 : h1;
        lstm_step<<<dim3(HID / 32, BATCH / 32), 128, 0, stream>>>(
            Whb, proj, x, t, hp, hn, cbuf, hfb);
    }

    final_proj<<<(BATCH * NCLS + 255) / 256, 256, 0, stream>>>(hfb, Wph, bp, (float*)d_out);
}
